// PLCnnDTM_i_19816979104358
// MI455X (gfx1250) — matrix operand order repaired and checked by reading, not compiled, not run on hardware
//
#include <hip/hip_runtime.h>
#include <hip/hip_bf16.h>

// ---------- types for WMMA (gfx1250, wave32) ----------
typedef __bf16 bf16;
typedef bf16  bf16x8  __attribute__((ext_vector_type(8)));
typedef bf16  bf16x16 __attribute__((ext_vector_type(16)));
typedef float f32x8   __attribute__((ext_vector_type(8)));

#define BATCH 1024
#define PPX   784        // 28*28
#define ZSTR  864        // 848 concat width padded to K multiple of 32
#define NF1   64         // fc1 output width
#define NCHUNK 27        // 864 / 32 K-chunks

__device__ __forceinline__ unsigned short f32_to_bf16_bits(float f) {
    unsigned int u = __float_as_uint(f);
    unsigned int r = (u + 0x7FFFu + ((u >> 16) & 1u)) >> 16;   // RNE
    return (unsigned short)r;
}
__device__ __forceinline__ float bf16_bits_to_f32(unsigned short u) {
    return __uint_as_float(((unsigned int)u) << 16);
}

// ---------------------------------------------------------------------------
// Kernel 1: fc1_w [848,64] f32 -> bf16 in per-lane WMMA fragment order:
//   bswz[((c*4 + ntile)*32 + lane)*16 + j]  (c = K-chunk of 32)
// where lane holds col = ntile*16 + (lane&15), K = c*32 + (lane>>4)*8 + j (j<8), c*32 + 16 + (lane>>4)*8 + (j-8) (j>=8).
// Rows >= 848 are zero padding.
// ---------------------------------------------------------------------------
__global__ void prep_fc1w_kernel(const float* __restrict__ fc1w,
                                 unsigned short* __restrict__ bswz) {
    int idx = blockIdx.x * blockDim.x + threadIdx.x;   // 27*4*32*16 = 55296 threads
    int j    = idx & 15;
    int lane = (idx >> 4) & 31;
    int i    = (idx >> 9) & 3;
    int c    = idx >> 11;
    int k = c * 32 + (lane >> 4) * 8 + (j < 8 ? j : 16 + (j - 8));
    int n = i * 16 + (lane & 15);
    float v = (k < 848) ? fc1w[k * NF1 + n] : 0.0f;
    bswz[idx] = f32_to_bf16_bits(v);
}

// ---------------------------------------------------------------------------
// Kernel 2: persistence landscape (top-2 tents over 32 t-steps) + tiny dense
// grid = (1024, 2) : blockIdx.y selects branch. 256 threads (8 wave32).
// ---------------------------------------------------------------------------
__global__ void landscape_kernel(const float* __restrict__ dtm1,
                                 const float* __restrict__ dtm2,
                                 const float* __restrict__ g1w, const float* __restrict__ g1b,
                                 const float* __restrict__ g2w, const float* __restrict__ g2b,
                                 unsigned short* __restrict__ zb) {
    const int b      = blockIdx.x;
    const int branch = blockIdx.y;
    const float* v  = (branch == 0 ? dtm1 : dtm2) + (size_t)b * PPX;
    const float  t0 = branch == 0 ? 0.01f : 0.05f;
    const float  dt = branch == 0 ? (0.29f - 0.01f) / 31.0f : (0.30f - 0.05f) / 31.0f;
    const float* gw = branch == 0 ? g1w : g2w;
    const float* gb = branch == 0 ? g1b : g2b;

    __shared__ float sv[PPX];
    __shared__ float sred[256];
    __shared__ float feat[64];

    const int t = threadIdx.x;

    float pm = -1e30f;
    for (int i = t; i < PPX; i += 256) { float x = v[i]; sv[i] = x; pm = fmaxf(pm, x); }
    sred[t] = pm;
    __syncthreads();
    for (int s = 128; s > 0; s >>= 1) {
        if (t < s) sred[t] = fmaxf(sred[t], sred[t + s]);
        __syncthreads();
    }
    const float death = sred[0];

    const int wave = t >> 5, lane = t & 31;
    for (int ti = wave; ti < 32; ti += 8) {
        const float tv = t0 + dt * (float)ti;
        float m1 = -1e30f, m2 = -1e30f;
        for (int i = lane; i < PPX; i += 32) {
            float tent = fminf(tv - sv[i], death - tv);
            tent = fmaxf(tent, 0.0f);
            if (tent > m1) { m2 = m1; m1 = tent; }
            else           { m2 = fmaxf(m2, tent); }
        }
        #pragma unroll
        for (int off = 16; off > 0; off >>= 1) {
            float o1 = __shfl_xor(m1, off, 32);
            float o2 = __shfl_xor(m2, off, 32);
            float hi = fmaxf(m1, o1), lo = fminf(m1, o1);
            m2 = fmaxf(lo, fmaxf(m2, o2));
            m1 = hi;
        }
        if (lane == 0) { feat[ti] = m1; feat[32 + ti] = m2; }   // [k*32 + t]
    }
    __syncthreads();

    if (t < 32) {
        float acc = gb[t];
        #pragma unroll 8
        for (int i = 0; i < 64; i++) acc += feat[i] * gw[i * 32 + t];
        acc = fmaxf(acc, 0.0f);
        zb[(size_t)b * ZSTR + PPX + branch * 32 + t] = f32_to_bf16_bits(acc);
    }
    if (branch == 0 && t >= 64 && t < 80)
        zb[(size_t)b * ZSTR + 848 + (t - 64)] = 0;
}

// ---------------------------------------------------------------------------
// Kernel 3: fused conv(1->32,3x3 SAME)+ReLU, conv(32->1,3x3 SAME)+ReLU
// ---------------------------------------------------------------------------
__global__ void conv_fused_kernel(const float* __restrict__ xg,
                                  const float* __restrict__ w1, const float* __restrict__ b1,
                                  const float* __restrict__ w2, const float* __restrict__ b2v,
                                  unsigned short* __restrict__ zb) {
    const int b = blockIdx.x;
    __shared__ float          xin[30 * 30];
    __shared__ float          w1s[288];
    __shared__ float          w2s[288];
    __shared__ float          b1s[32];
    __shared__ unsigned short h1[PPX * 32];

    const int t = threadIdx.x;
    __builtin_prefetch(xg + (size_t)b * PPX, 0, 0);

    for (int i = t; i < 900; i += 256) xin[i] = 0.0f;
    if (t < 288) { w1s[t] = w1[t]; w2s[t] = w2[t]; }
    if (t < 32)  b1s[t] = b1[t];
    __syncthreads();
    for (int i = t; i < PPX; i += 256) {
        int y = i / 28, x = i % 28;
        xin[(y + 1) * 30 + (x + 1)] = xg[(size_t)b * PPX + i];
    }
    __syncthreads();

    for (int idx = t; idx < PPX * 32; idx += 256) {
        int c = idx & 31, p = idx >> 5;
        int y = p / 28, x = p % 28;
        float acc = b1s[c];
        #pragma unroll
        for (int dy = 0; dy < 3; dy++)
            #pragma unroll
            for (int dx = 0; dx < 3; dx++)
                acc += xin[(y + dy) * 30 + (x + dx)] * w1s[(dy * 3 + dx) * 32 + c];
        h1[idx] = f32_to_bf16_bits(fmaxf(acc, 0.0f));
    }
    __syncthreads();

    const float b2 = b2v[0];
    for (int p = t; p < PPX; p += 256) {
        int y = p / 28, x = p % 28;
        float acc = b2;
        for (int dy = 0; dy < 3; dy++) {
            int yy = y + dy - 1;
            if (yy < 0 || yy >= 28) continue;
            for (int dx = 0; dx < 3; dx++) {
                int xx = x + dx - 1;
                if (xx < 0 || xx >= 28) continue;
                const unsigned short* hp = &h1[(yy * 28 + xx) * 32];
                const float*          wp = &w2s[(dy * 3 + dx) * 32];
                #pragma unroll
                for (int c = 0; c < 32; c++) acc += bf16_bits_to_f32(hp[c]) * wp[c];
            }
        }
        zb[(size_t)b * ZSTR + p] = f32_to_bf16_bits(fmaxf(acc, 0.0f));
    }
}

// ---------------------------------------------------------------------------
// Kernel 4: fc1 via v_wmma_f32_16x16x32_bf16 with async double-buffered B.
// B (pre-swizzled fragments) is pulled one 4KB K-chunk ahead into LDS with
// global_load_async_to_lds_b128 (ASYNCcnt), consumed via ds_load_b128.
// 8 waves / block, each wave owns 16(M) x 64(N); grid = 8 blocks.
// ---------------------------------------------------------------------------
__global__ void fc1_wmma_kernel(const unsigned short* __restrict__ zb_u,
                                const unsigned short* __restrict__ bswz,
                                const float* __restrict__ fc1b,
                                float* __restrict__ a1) {
    __shared__ __align__(16) unsigned short bstage[2][2048];   // 2 x 4KB

    const bf16* zb = reinterpret_cast<const bf16*>(zb_u);
    const int t     = threadIdx.x;
    const int wave  = t >> 5;
    const int lane  = t & 31;
    const int m0    = (blockIdx.x * 8 + wave) * 16;
    const int khalf = lane >> 4;
    const int lrow  = lane & 15;

    // prologue: async-stage B chunk 0 (each thread moves 16B; block moves 4KB)
    {
        unsigned long long ga = (unsigned long long)(const void*)(bswz + t * 8);
        unsigned ldsa = (unsigned)(unsigned long long)(const void*)(&bstage[0][t * 8]);
        asm volatile("global_load_async_to_lds_b128 %0, %1, off"
                     :: "v"(ldsa), "v"(ga) : "memory");
    }

    f32x8 acc[4] = {f32x8{}, f32x8{}, f32x8{}, f32x8{}};

    for (int c = 0; c < NCHUNK; c++) {
        if (c + 1 < NCHUNK) {
            // prefetch next chunk into the other buffer (safe: barrier at loop
            // end guarantees reads of chunk c-1 from this buffer are done)
            unsigned long long ga =
                (unsigned long long)(const void*)(bswz + (size_t)(c + 1) * 2048 + t * 8);
            unsigned ldsa =
                (unsigned)(unsigned long long)(const void*)(&bstage[(c + 1) & 1][t * 8]);
            asm volatile("global_load_async_to_lds_b128 %0, %1, off"
                         :: "v"(ldsa), "v"(ga) : "memory");
            asm volatile("s_wait_asynccnt 0x1" ::: "memory");   // chunk c landed
        } else {
            asm volatile("s_wait_asynccnt 0x0" ::: "memory");
        }
        __syncthreads();    // all waves' slices of chunk c visible

        const int kc = c * 32;
        // A fragment: row = m0+lrow; lane holds K = kc+khalf*8+{0..7} and +16
        const bf16* ap = zb + (size_t)(m0 + lrow) * ZSTR + kc + khalf * 8;
        bf16x8 alo = *reinterpret_cast<const bf16x8*>(ap);
        bf16x8 ahi = *reinterpret_cast<const bf16x8*>(ap + 16);
        bf16x16 amat = __builtin_shufflevector(alo, ahi,
            0, 1, 2, 3, 4, 5, 6, 7, 8, 9, 10, 11, 12, 13, 14, 15);

        // B fragments from LDS: tile i at [i*512 + lane*16], 32B per lane
        const unsigned short* bb = &bstage[c & 1][lane * 16];
        #pragma unroll
        for (int i = 0; i < 4; i++) {
            bf16x8 blo = *reinterpret_cast<const bf16x8*>(bb + i * 512);
            bf16x8 bhi = *reinterpret_cast<const bf16x8*>(bb + i * 512 + 8);
            bf16x16 bmat = __builtin_shufflevector(blo, bhi,
                0, 1, 2, 3, 4, 5, 6, 7, 8, 9, 10, 11, 12, 13, 14, 15);
            acc[i] = __builtin_amdgcn_wmma_f32_16x16x32_bf16(
                false, amat, false, bmat, (short)0, acc[i], false, false);
        }
        __syncthreads();    // protect this buffer from next prefetch overwrite
    }

    // C/D layout: lanes 0-15 hold M=r, lanes 16-31 hold M=r+8
    const int rbase = m0 + (khalf << 3);
    #pragma unroll
    for (int i = 0; i < 4; i++) {
        const int col  = i * 16 + lrow;
        const float bias = fc1b[col];
        #pragma unroll
        for (int r = 0; r < 8; r++) {
            float v = acc[i][r] + bias;
            a1[(size_t)(rbase + r) * NF1 + col] = fmaxf(v, 0.0f);
        }
    }
}

// ---------------------------------------------------------------------------
// Kernel 5: fc2 [1024,64] x [64,10] + bias  (tiny: per-thread dot)
// ---------------------------------------------------------------------------
__global__ void fc2_kernel(const float* __restrict__ a1,
                           const float* __restrict__ fc2w,
                           const float* __restrict__ fc2b,
                           float* __restrict__ out) {
    int idx = blockIdx.x * blockDim.x + threadIdx.x;
    if (idx >= BATCH * 10) return;
    int b = idx / 10, c = idx % 10;
    float acc = fc2b[c];
    #pragma unroll 8
    for (int i = 0; i < 64; i++) acc += a1[(size_t)b * 64 + i] * fc2w[i * 10 + c];
    out[idx] = acc;
}

// ---------------------------------------------------------------------------
extern "C" void kernel_launch(void* const* d_in, const int* in_sizes, int n_in,
                              void* d_out, int out_size, void* d_ws, size_t ws_size,
                              hipStream_t stream) {
    (void)in_sizes; (void)n_in; (void)out_size; (void)ws_size;

    const float* x     = (const float*)d_in[0];
    const float* dtm1  = (const float*)d_in[1];
    const float* dtm2  = (const float*)d_in[2];
    const float* w1    = (const float*)d_in[3];
    const float* b1    = (const float*)d_in[4];
    const float* w2    = (const float*)d_in[5];
    const float* b2    = (const float*)d_in[6];
    const float* g1w   = (const float*)d_in[7];
    const float* g1b   = (const float*)d_in[8];
    const float* g2w   = (const float*)d_in[9];
    const float* g2b   = (const float*)d_in[10];
    const float* fc1w  = (const float*)d_in[11];
    const float* fc1b  = (const float*)d_in[12];
    const float* fc2w  = (const float*)d_in[13];
    const float* fc2b  = (const float*)d_in[14];
    float* out = (float*)d_out;

    const size_t ZB_BYTES = (size_t)BATCH * ZSTR * sizeof(unsigned short); // 1,769,472
    const size_t WB_BYTES = (size_t)NCHUNK * 4 * 32 * 16 * sizeof(unsigned short); // 110,592
    unsigned short* zb   = (unsigned short*)d_ws;
    unsigned short* bswz = (unsigned short*)((char*)d_ws + ZB_BYTES);
    float*          a1   = (float*)((char*)d_ws + ZB_BYTES + WB_BYTES);

    prep_fc1w_kernel<<<(NCHUNK * 4 * 32 * 16) / 256, 256, 0, stream>>>(fc1w, bswz);
    landscape_kernel<<<dim3(BATCH, 2), 256, 0, stream>>>(dtm1, dtm2, g1w, g1b, g2w, g2b, zb);
    conv_fused_kernel<<<BATCH, 256, 0, stream>>>(x, w1, b1, w2, b2, zb);
    fc1_wmma_kernel<<<BATCH / 128, 256, 0, stream>>>(zb, bswz, fc1b, a1);
    fc2_kernel<<<(BATCH * 10 + 255) / 256, 256, 0, stream>>>(a1, fc2w, fc2b, out);
}
